// CarbonAwareLSTM_11003706212664
// MI455X (gfx1250) — matrix operand order repaired and checked by reading, not compiled, not run on hardware
//
#include <hip/hip_runtime.h>
#include <hip/hip_bf16.h>
#include <stdint.h>

// ---------------- problem constants ----------------
#define BB  64            // batch
#define TT  4096          // timesteps
#define FF  64            // input features
#define UU  128           // units
#define G4  512           // 4*U (gates i,f,g,o)
#define KK  192           // F + U : fused GEMM depth  [x_t | h]
#define KT  6             // KK / 32 (bf16 WMMA K=32)
#define WGM 16            // batch rows per workgroup (WMMA M)
#define NWG 4             // BB / WGM workgroups, independent recurrences
#define NTH 256           // 8 waves: wave w owns unit slice [16w,16w+16) for all 4 gates

// ---------------- LDS layout (bytes) ----------------
// Row stride 200 halves (400B): 36n mod 64 is a permutation of 4-bank groups
// -> conflict-free ds_load_b128 B-fragment fetches across 16 lanes.
#define WT_STRIDE 200
#define WT_BYTES  (G4 * WT_STRIDE * 2)       // 204800 : [W;R]^T as bf16, col-major
#define AP_OFF    WT_BYTES
#define AP_BYTES  (WGM * WT_STRIDE * 2)      // 6400   : activation panel [x_t | h]
#define XS_OFF    (AP_OFF + AP_BYTES)        // 211200 : TDM staging for x_{t+1} (f32)
#define XS_BYTES  (WGM * FF * 4)             // 4096
#define SMEM_BYTES (XS_OFF + XS_BYTES)       // 215296  (< 320KB WGP LDS)

typedef __attribute__((ext_vector_type(8)))  __bf16       v8bf;
typedef __attribute__((ext_vector_type(16))) __bf16       v16bf;
typedef __attribute__((ext_vector_type(8)))  float        v8f;
typedef __attribute__((ext_vector_type(4)))  unsigned int v4u;
typedef __attribute__((ext_vector_type(8)))  int          v8i;
typedef __attribute__((ext_vector_type(4)))  int          v4i;

#if __has_builtin(__builtin_amdgcn_tensor_load_to_lds) && \
    __has_builtin(__builtin_amdgcn_s_wait_tensorcnt)
#define USE_TDM 1
#else
#define USE_TDM 0
#endif

__device__ __forceinline__ float sigmoid_f(float x) {
  return 1.0f / (1.0f + __expf(-x));
}
__device__ __forceinline__ float tanh_f(float x) {
  // 1 - 2/(e^{2x}+1); saturates correctly at +/-1 via inf/0 of __expf
  return 1.0f - 2.0f / (__expf(2.0f * x) + 1.0f);
}

// 16 contiguous bf16 (32B) as two 16B-aligned b128 loads
__device__ __forceinline__ v16bf ld_frag(const __bf16* p) {
  v8bf lo = *(const v8bf*)(p);
  v8bf hi = *(const v8bf*)(p + 8);
  return __builtin_shufflevector(lo, hi, 0,1,2,3,4,5,6,7,8,9,10,11,12,13,14,15);
}
// A-fragment: two disjoint 8-half runs per ISA A layout (K+0..7, K+16..23)
__device__ __forceinline__ v16bf ld_frag2(const __bf16* p0, const __bf16* p1) {
  v8bf lo = *(const v8bf*)(p0);
  v8bf hi = *(const v8bf*)(p1);
  return __builtin_shufflevector(lo, hi, 0,1,2,3,4,5,6,7,8,9,10,11,12,13,14,15);
}

#if USE_TDM
// TDM descriptor per CDNA5 ISA:
// 2-D tile: tile_dim0=64 (f32 elems), tile_dim1=16 rows, row stride T*F elems.
// group0: count=1 | lds_addr | global_addr(57b) | type=2.
// Toolchain is the clang-23/therock lane: 6-arg builtin form
//   (uint32x4 g0, int32x8 g1, int32x4 g2, int32x4 g3, int32x8, i32 cpol)
__device__ __forceinline__ void tdm_load_x_tile(uint64_t gaddr_bytes) {
  v4u g0;
  g0[0] = 1u;                                   // count=1, user descriptor
  g0[1] = (unsigned)XS_OFF;                     // lds_addr (bytes)
  g0[2] = (unsigned)(gaddr_bytes & 0xffffffffu);
  g0[3] = (unsigned)((gaddr_bytes >> 32) & 0x01ffffffu) | 0x80000000u; // type=2
  v8i g1;
  g1[0] = 0x00020000;        // wg_mask=0 (not in cluster), data_size=2 -> 4B
  g1[1] = (int)(FF << 16);   // tensor_dim0 = 64   (bits 79:48, lo half here)
  g1[2] = (int)(WGM << 16);  // tensor_dim1 = 16   (bits 111:80, lo half here)
  g1[3] = (int)(FF << 16);   // tile_dim0  = 64    (bits 127:112)
  g1[4] = WGM;               // tile_dim1  = 16, tile_dim2 = 0
  g1[5] = TT * FF;           // tensor_dim0_stride = 262144 elems (lo32)
  g1[6] = 0;                 // stride hi16, dim1_stride lo16
  g1[7] = 0;
  v4i z4 = {0, 0, 0, 0};                 // groups 2/3 unused (2-D tensor)
  v8i z8 = {0, 0, 0, 0, 0, 0, 0, 0};     // extra operand of 6-arg form, unused
  __builtin_amdgcn_tensor_load_to_lds(g0, g1, z4, z4, z8, 0);
}
#endif

__global__ void __launch_bounds__(NTH, 1)
lstm_persistent_kernel(const float* __restrict__ x,
                       const float* __restrict__ wk,   // [F,4U]
                       const float* __restrict__ wr,   // [U,4U]
                       const float* __restrict__ bias, // [4U]
                       float* __restrict__ out)        // [B,U]
{
  extern __shared__ __align__(16) char smem[];
  __bf16* Wt = (__bf16*)(smem);            // [512][200] bf16, Wt[n][k] = W[k][n]
  __bf16* Ap = (__bf16*)(smem + AP_OFF);   // [16][200] bf16, [x_t(64) | h(128)]
  float*  Xs = (float*)(smem + XS_OFF);    // [16][64] f32 TDM staging

  const int  tid  = threadIdx.x;
  const int  wave = tid >> 5;
  const int  lane = tid & 31;
  const int  ln   = lane & 15;
  const bool hiH  = lane >= 16;
  const int  wg   = blockIdx.x;

  // ---- one-time: stage [kernel; recurrent]^T as bf16 into LDS ----
  for (int idx = tid; idx < KK * G4; idx += NTH) {
    int k = idx >> 9;                 // 0..191   (coalesced over n)
    int n = idx & (G4 - 1);           // 0..511
    float w = (k < FF) ? wk[k * G4 + n] : wr[(k - FF) * G4 + n];
    Wt[n * WT_STRIDE + k] = (__bf16)w;
  }
  // zero activation panel: h_0 = 0 (x cols overwritten below)
  for (int idx = tid; idx < WGM * WT_STRIDE; idx += NTH) Ap[idx] = (__bf16)0.0f;

  // per-lane bias for this wave's unit slice (C/D layout: N = lane%16)
  const int   u  = wave * 16 + ln;
  const float b_i = bias[0 * UU + u];
  const float b_f = bias[1 * UU + u];
  const float b_g = bias[2 * UU + u];
  const float b_o = bias[3 * UU + u];

  const uint64_t xbase =
      (uint64_t)(uintptr_t)x + (uint64_t)wg * WGM * TT * FF * 4ull;

  // ---- x_0 into the panel ----
#if USE_TDM
  if (wave == 0) {
    tdm_load_x_tile(xbase);
    __builtin_amdgcn_s_wait_tensorcnt(0);
  }
  __syncthreads();
  for (int idx = tid; idx < WGM * FF; idx += NTH)
    Ap[(idx >> 6) * WT_STRIDE + (idx & 63)] = (__bf16)Xs[idx];
#else
  for (int idx = tid; idx < WGM * FF; idx += NTH) {
    int m = idx >> 6, f = idx & 63;
    Ap[m * WT_STRIDE + f] =
        (__bf16)x[(uint64_t)(wg * WGM + m) * TT * FF + f];
  }
#endif
  __syncthreads();

  // B-fragment bases: wave w owns columns {u, U+u, 2U+u, 3U+u} => i,f,g,o
  const __bf16* Bi = Wt + (0 * UU + u) * WT_STRIDE;
  const __bf16* Bf = Wt + (1 * UU + u) * WT_STRIDE;
  const __bf16* Bg = Wt + (2 * UU + u) * WT_STRIDE;
  const __bf16* Bo = Wt + (3 * UU + u) * WT_STRIDE;
  const __bf16* Arow = Ap + ln * WT_STRIDE;   // A layout: lane holds row M=lane%16

  v8f cst = {0.f, 0.f, 0.f, 0.f, 0.f, 0.f, 0.f, 0.f};  // cell state, registers

  for (int t = 0; t < TT; ++t) {
#if USE_TDM
    if (wave == 0 && t + 1 < TT)                    // prefetch x_{t+1} via TDM,
      tdm_load_x_tile(xbase + (uint64_t)(t + 1) * (FF * 4)); // overlaps WMMAs
#endif
    v8f ai = {b_i, b_i, b_i, b_i, b_i, b_i, b_i, b_i};
    v8f af = {b_f, b_f, b_f, b_f, b_f, b_f, b_f, b_f};
    v8f ag = {b_g, b_g, b_g, b_g, b_g, b_g, b_g, b_g};
    v8f ao = {b_o, b_o, b_o, b_o, b_o, b_o, b_o, b_o};

#pragma unroll
    for (int kt = 0; kt < KT; ++kt) {
      const int ka = kt * 32 + (hiH ? 16 : 0);      // A: 16 contiguous K
      v16bf a = ld_frag(Arow + ka);
      const int kb = kt * 32 + (hiH ? 16 : 0);      // B: 16 contiguous K
      v16bf vbi = ld_frag(Bi + kb);
      v16bf vbf = ld_frag(Bf + kb);
      v16bf vbg = ld_frag(Bg + kb);
      v16bf vbo = ld_frag(Bo + kb);
      ai = __builtin_amdgcn_wmma_f32_16x16x32_bf16(false, a, false, vbi, (short)0, ai, false, false);
      af = __builtin_amdgcn_wmma_f32_16x16x32_bf16(false, a, false, vbf, (short)0, af, false, false);
      ag = __builtin_amdgcn_wmma_f32_16x16x32_bf16(false, a, false, vbg, (short)0, ag, false, false);
      ao = __builtin_amdgcn_wmma_f32_16x16x32_bf16(false, a, false, vbo, (short)0, ao, false, false);
    }
    __syncthreads();   // all Ap reads of step t complete before h_t overwrite

    // Gates + state update entirely in registers (i,f,g,o share (m,u) mapping).
#pragma unroll
    for (int r = 0; r < 8; ++r) {                   // C layout: M=r (+8 upper half)
      float gi = sigmoid_f(ai[r]);
      float gf = sigmoid_f(af[r]);
      float gg = tanh_f(ag[r]);
      float go = sigmoid_f(ao[r]);
      float c  = gf * cst[r] + gi * gg;
      cst[r]   = c;
      float h  = go * tanh_f(c);
      int m = hiH ? (r + 8) : r;
      if (t == TT - 1) {
        out[(uint64_t)(wg * WGM + m) * UU + u] = h; // final hidden state, f32
      } else {
        Ap[m * WT_STRIDE + FF + u] = (__bf16)h;     // h_t -> panel for step t+1
      }
    }

    if (t + 1 < TT) {
#if USE_TDM
      if (wave == 0) {                              // consume prefetched x_{t+1}
        __builtin_amdgcn_s_wait_tensorcnt(0);
        for (int idx = lane; idx < WGM * FF; idx += 32)
          Ap[(idx >> 6) * WT_STRIDE + (idx & 63)] = (__bf16)Xs[idx];
      }
#else
      for (int idx = tid; idx < WGM * FF; idx += NTH) {
        int m = idx >> 6, f = idx & 63;
        Ap[m * WT_STRIDE + f] =
            (__bf16)x[((uint64_t)(wg * WGM + m) * TT + (t + 1)) * FF + f];
      }
#endif
    }
    __syncthreads();
  }
}

extern "C" void kernel_launch(void* const* d_in, const int* in_sizes, int n_in,
                              void* d_out, int out_size, void* d_ws, size_t ws_size,
                              hipStream_t stream) {
  (void)in_sizes; (void)n_in; (void)out_size; (void)d_ws; (void)ws_size;
  const float* x  = (const float*)d_in[0];  // [64,4096,64]
  const float* wk = (const float*)d_in[1];  // [64,512]
  const float* wr = (const float*)d_in[2];  // [128,512]
  const float* bs = (const float*)d_in[3];  // [512]
  float* out = (float*)d_out;               // [64,128]

  // >64KB dynamic LDS needs an opt-in; host-side, idempotent, capture-safe.
  (void)hipFuncSetAttribute((const void*)lstm_persistent_kernel,
                            hipFuncAttributeMaxDynamicSharedMemorySize,
                            SMEM_BYTES);

  lstm_persistent_kernel<<<dim3(NWG), dim3(NTH), SMEM_BYTES, stream>>>(
      x, wk, wr, bs, out);
}
